// DGCNN_89146341196038
// MI455X (gfx1250) — matrix operand order repaired and checked by reading, not compiled, not run on hardware
//
#include <hip/hip_runtime.h>

// ---------------------------------------------------------------------------
// DGCNN forward for MI455X (gfx1250, wave32).
//  - All 1x1-conv GEMMs on v_wmma_f32_16x16x32_bf16 (bf16 in, f32 accumulate).
//  - Weights pre-packed into A-fragment order (contiguous 32B per lane).
//  - Inter-conv activations stored column-major bf16 -> B-fragment K-step is
//    one contiguous v16bf load. One wave: 16 columns, loops all Co tiles.
//  - All shapes compile-time: address math is 32-bit with constant strides.
//  - KNN / BN / max are VALU kernels; reductions deterministic (no atomics).
// ---------------------------------------------------------------------------

typedef __bf16 bf16_t;
typedef __attribute__((ext_vector_type(16))) bf16_t v16bf;
typedef __attribute__((ext_vector_type(8)))  float  v8f;

#define BATCH 16
#define NPTS  2048
#define KNBR  20
#define EMB   1024
#define SLOPE 0.2f
#define EPSB  1e-5f

__device__ __forceinline__ bf16_t f2bf(float f) { return (bf16_t)f; }
__device__ __forceinline__ unsigned short f2bf_bits(float f) {
    return __builtin_bit_cast(unsigned short, (bf16_t)f);
}
__device__ __forceinline__ int imin(int a, int b) { return a < b ? a : b; }
__device__ __forceinline__ int imax(int a, int b) { return a > b ? a : b; }

// ------------------------- squared norms ------------------------------------
__global__ void k_sqnorm(const float* __restrict__ F, float* __restrict__ xx,
                         int C, int Nn) {
    int i = blockIdx.x * blockDim.x + threadIdx.x;      // over B*N
    int b = i / Nn, n = i % Nn;
    const float* f = F + b * C * Nn + n;
    float s = 0.f;
    for (int c = 0; c < C; ++c) { float v = f[c * Nn]; s += v * v; }
    xx[i] = s;
}

// ------------------------------ KNN -----------------------------------------
template <int C>
__global__ void k_knn(const float* __restrict__ F, const float* __restrict__ xx,
                      int* __restrict__ idx, int Nn, int Kk) {
    const int TS = 128;
    __shared__ float tile[TS * C];
    __shared__ float txx[TS];
    int tid = threadIdx.x;                      // blockDim = 128
    int n   = blockIdx.x * TS + tid;
    int b   = blockIdx.y;
    const float* Fb = F + b * C * Nn;

    float cf[C];
#pragma unroll
    for (int c = 0; c < C; ++c) cf[c] = Fb[c * Nn + n];

    float best[KNBR]; int bidx[KNBR];
#pragma unroll
    for (int t = 0; t < KNBR; ++t) { best[t] = 3.4e38f; bidx[t] = 0; }

    for (int m0 = 0; m0 < Nn; m0 += TS) {
        __syncthreads();
        for (int u = tid; u < TS * C; u += 128) {
            int c = u / TS, mm = u % TS;
            tile[mm * C + c] = Fb[c * Nn + m0 + mm];
        }
        if (tid < TS) txx[tid] = xx[b * Nn + m0 + tid];
        __syncthreads();
        for (int mm = 0; mm < TS; ++mm) {
            float dot = 0.f;
#pragma unroll
            for (int c = 0; c < C; ++c) dot += cf[c] * tile[mm * C + c];
            float s = txx[mm] - 2.f * dot;      // ranking == reference top_k
            if (s < best[KNBR - 1]) {
                int p = KNBR - 1;
                while (p > 0 && best[p - 1] > s) {
                    best[p] = best[p - 1]; bidx[p] = bidx[p - 1]; --p;
                }
                best[p] = s; bidx[p] = m0 + mm;
            }
        }
    }
    int* op = idx + (b * Nn + n) * Kk;
    for (int t = 0; t < Kk; ++t) op[t] = bidx[t];
}

// --------------- pack f32 weights into WMMA A-fragment order ----------------
// Wp[((rt*KS + ks)*32 + lane)*16 + e]; e=2v(+1) <-> K = ks*32 + ka(v,hh)(+1).
__global__ void k_wpack(const float* __restrict__ W, unsigned short* __restrict__ Wp,
                        int Ci, int Co) {
    int ksteps = (Ci + 31) >> 5;
    int total  = (Co >> 4) * ksteps * 32 * 16;
    int i = blockIdx.x * blockDim.x + threadIdx.x;
    if (i >= total) return;
    int e    = i & 15;
    int lane = (i >> 4) & 31;
    int ks   = (i >> 9) % ksteps;
    int rt   = (i >> 9) / ksteps;
    int hh = lane >> 4, l16 = lane & 15;
    int v = e >> 1, odd = e & 1;
    int ka = 2 * v + 16 * hh;
    int c  = (ks << 5) + ka + odd;
    int row = rt * 16 + l16;
    float wv = (c < Ci && row < Co) ? W[row * Ci + c] : 0.f;
    Wp[i] = f2bf_bits(wv);
}

// -------- branchless graph-feature element (clamped loads, cndmask) ---------
template <int C, int Nn>
__device__ __forceinline__ float gf_el(const float* __restrict__ Fb,
                                       int c, int n, int m) {
    constexpr int Ci = 2 * C;
    int cd = imin(c, C - 1);
    int cc = imin(imax(c - C, 0), C - 1);
    float dif = Fb[cd * Nn + m] - Fb[cd * Nn + n];
    float cen = Fb[cc * Nn + n];
    float r = (c < C) ? dif : cen;
    return (c < Ci) ? r : 0.f;
}

// --------- WMMA conv over gathered graph features: Y = W * gf(F, idx) -------
// One wave = 16 columns; B-fragments loaded once, loop over all Co tiles.
template <int C, int Co, int Nn>
__global__ void k_conv_gf_t(const float* __restrict__ F, const int* __restrict__ idx,
                            const unsigned short* __restrict__ Wp,
                            float* __restrict__ Y) {
    constexpr int Ci = 2 * C;
    constexpr int KS = (Ci + 31) / 32;
    constexpr int RT = Co / 16;
    constexpr int L  = Nn * KNBR;
    int wave = (blockIdx.x * blockDim.x + threadIdx.x) >> 5;  // column tile (grid exact)
    int lane = threadIdx.x & 31;
    int hh = lane >> 4, l16 = lane & 15;
    int j0 = wave * 16;
    int b  = j0 / L, jl = j0 % L;
    int jcol = jl + l16;
    int n  = jcol / KNBR;
    int kk = jcol % KNBR;
    int mIdx = idx[(b * Nn + n) * KNBR + kk];
    const float* Fb = F + b * (C * Nn);

    v16bf bm[KS];
#pragma unroll
    for (int ks = 0; ks < KS; ++ks) {
#pragma unroll
        for (int v = 0; v < 8; ++v) {
            int cb = (ks << 5) + 2 * v + 16 * hh;
            bm[ks][2 * v]     = f2bf(gf_el<C, Nn>(Fb, cb,     n, mIdx));
            bm[ks][2 * v + 1] = f2bf(gf_el<C, Nn>(Fb, cb + 1, n, mIdx));
        }
    }
    float* Yb = Y + (unsigned)(b * (Co * L) + hh * (8 * L) + jcol);
#pragma unroll
    for (int rt = 0; rt < RT; ++rt) {
        v8f acc = {};
#pragma unroll
        for (int ks = 0; ks < KS; ++ks) {
            v16bf a = *(const v16bf*)(Wp + ((rt * KS + ks) * 32 + lane) * 16);
            acc = __builtin_amdgcn_wmma_f32_16x16x32_bf16(
                false, a, false, bm[ks], (short)0, acc, false, false);
        }
#pragma unroll
        for (int r = 0; r < 8; ++r)
            Yb[(rt * 16 + r) * L] = acc[r];
    }
}

// --------------- WMMA conv over dense bf16 activations ----------------------
// Act column-major [B, L, Ci] bf16-bits; one v16bf load per B K-step.
template <int Ci, int Co, int L>
__global__ void k_conv_bf_t(const unsigned short* __restrict__ Act,
                            const unsigned short* __restrict__ Wp,
                            float* __restrict__ Y) {
    static_assert(Ci % 32 == 0, "Ci must be a multiple of 32");
    constexpr int KS = Ci / 32;
    constexpr int RT = Co / 16;
    int wave = (blockIdx.x * blockDim.x + threadIdx.x) >> 5;  // column tile (grid exact)
    int lane = threadIdx.x & 31;
    int hh = lane >> 4, l16 = lane & 15;
    int j0 = wave * 16;
    int b  = j0 / L, jl = j0 % L;
    int jcol = jl + l16;
    const unsigned short* Ab = Act + (unsigned)(b * (L * Ci) + jcol * Ci + hh * 16);

    v16bf bm[KS];
#pragma unroll
    for (int ks = 0; ks < KS; ++ks)
        bm[ks] = *(const v16bf*)(Ab + ks * 32);               // 16 contiguous K

    float* Yb = Y + (unsigned)(b * (Co * L) + hh * (8 * L) + jcol);
    for (int rt = 0; rt < RT; ++rt) {
        v8f acc = {};
#pragma unroll
        for (int ks = 0; ks < KS; ++ks) {
            v16bf a = *(const v16bf*)(Wp + ((rt * KS + ks) * 32 + lane) * 16);
            acc = __builtin_amdgcn_wmma_f32_16x16x32_bf16(
                false, a, false, bm[ks], (short)0, acc, false, false);
        }
#pragma unroll
        for (int r = 0; r < 8; ++r)
            Yb[(rt * 16 + r) * L] = acc[r];
    }
}

// ---------------- BN stats: per-(b,o) partials then finalize ----------------
__global__ void k_stats_part(const float* __restrict__ Y, float* __restrict__ part,
                             int L) {
    const float* p = Y + (size_t)blockIdx.x * L;        // blockIdx = b*Co + o
    float s = 0.f, q = 0.f;
    for (int i = threadIdx.x; i < L; i += blockDim.x) { float v = p[i]; s += v; q += v * v; }
    __shared__ float sh[512];
    sh[threadIdx.x] = s; sh[256 + threadIdx.x] = q; __syncthreads();
    for (int st = 128; st > 0; st >>= 1) {
        if ((int)threadIdx.x < st) {
            sh[threadIdx.x]       += sh[threadIdx.x + st];
            sh[256 + threadIdx.x] += sh[256 + threadIdx.x + st];
        }
        __syncthreads();
    }
    if (threadIdx.x == 0) { part[blockIdx.x * 2] = sh[0]; part[blockIdx.x * 2 + 1] = sh[256]; }
}

__global__ void k_stats_fin(const float* __restrict__ part, float* __restrict__ mu,
                            float* __restrict__ rs, int Co, int Bc, float invCnt) {
    int o = blockIdx.x * blockDim.x + threadIdx.x;
    if (o >= Co) return;
    float s = 0.f, q = 0.f;
    for (int b = 0; b < Bc; ++b) {
        s += part[(b * Co + o) * 2];
        q += part[(b * Co + o) * 2 + 1];
    }
    float m = s * invCnt;
    float var = q * invCnt - m * m;
    mu[o] = m;
    rs[o] = rsqrtf(var + EPSB);
}

// ------ BN + LeakyReLU, transpose to column-major bf16 [B, L, Ci] -----------
// Thread per column: coalesced reads per channel, vector (uint4) stores.
template <int Ci>
__global__ void k_bn_tr(const float* __restrict__ Y, unsigned short* __restrict__ Z,
                        const float* __restrict__ mu, const float* __restrict__ rs,
                        const float* __restrict__ g, const float* __restrict__ bb,
                        int L) {
    int i = blockIdx.x * blockDim.x + threadIdx.x;      // over B*L
    int b = i / L, pos = i % L;
    const float* Yb = Y + (size_t)b * Ci * L + pos;
    unsigned short zc[Ci];
#pragma unroll
    for (int o = 0; o < Ci; ++o) {
        float v = (Yb[(size_t)o * L] - mu[o]) * rs[o] * g[o] + bb[o];
        v = v >= 0.f ? v : SLOPE * v;
        zc[o] = f2bf_bits(v);
    }
    uint4* dst = (uint4*)(Z + (size_t)i * Ci);
#pragma unroll
    for (int q = 0; q < Ci / 8; ++q) dst[q] = ((const uint4*)zc)[q];
}

// -------- BN + LeakyReLU + max over k -> x (f32) + bf16 concat slice --------
// Xcat column-major [B, N, catC].
__global__ void k_bnmax(const float* __restrict__ Y, float* __restrict__ Xf,
                        unsigned short* __restrict__ Xcat,
                        const float* __restrict__ mu, const float* __restrict__ rs,
                        const float* __restrict__ g, const float* __restrict__ bb,
                        int Co, int Nn, int Kk, int catC, int catOff) {
    int i = blockIdx.x * blockDim.x + threadIdx.x;      // over B*Co*Nn
    int n = i % Nn; int o = (i / Nn) % Co; int b = i / (Nn * Co);
    const float* p = Y + ((size_t)(b * Co + o) * Nn + n) * Kk;
    float sc = rs[o] * g[o];
    float sh = bb[o] - mu[o] * sc;
    float mx = -3.4e38f;
    for (int t = 0; t < Kk; ++t) {
        float v = p[t] * sc + sh;
        v = v >= 0.f ? v : SLOPE * v;
        mx = fmaxf(mx, v);
    }
    Xf[(b * Co + o) * Nn + n] = mx;
    Xcat[(b * Nn + n) * catC + catOff + o] = f2bf_bits(mx);
}

// ---- final BN + LeakyReLU + LDS-tiled transpose [B,EMB,N] -> [B,N,EMB] -----
__global__ void k_final_t(const float* __restrict__ Y, float* __restrict__ Out,
                          const float* __restrict__ mu, const float* __restrict__ rs,
                          const float* __restrict__ g, const float* __restrict__ bb,
                          int Co, int Nn) {
    __shared__ float t[64][65];
    int b  = blockIdx.z;
    int o0 = blockIdx.y * 64, n0 = blockIdx.x * 64;
    int tx = threadIdx.x & 63;
    int ty = threadIdx.x >> 6;                          // 0..3
    const float* Yb = Y + (size_t)b * Co * Nn;
    for (int r = ty; r < 64; r += 4) {
        int o = o0 + r;
        float v = (Yb[(size_t)o * Nn + n0 + tx] - mu[o]) * rs[o] * g[o] + bb[o];
        v = v >= 0.f ? v : SLOPE * v;
        t[r][tx] = v;
    }
    __syncthreads();
    float* Ob = Out + (size_t)b * Nn * Co;
    for (int r = ty; r < 64; r += 4) {
        Ob[(size_t)(n0 + r) * Co + o0 + tx] = t[tx][r];
    }
}

// ---------------------------------------------------------------------------
extern "C" void kernel_launch(void* const* d_in, const int* in_sizes, int n_in,
                              void* d_out, int out_size, void* d_ws, size_t ws_size,
                              hipStream_t stream) {
    (void)in_sizes; (void)n_in; (void)out_size;
    const float* x  = (const float*)d_in[0];
    const float* W1 = (const float*)d_in[1];  const float* g1 = (const float*)d_in[2];  const float* b1 = (const float*)d_in[3];
    const float* W2 = (const float*)d_in[4];  const float* g2 = (const float*)d_in[5];  const float* b2 = (const float*)d_in[6];
    const float* W3 = (const float*)d_in[7];  const float* g3 = (const float*)d_in[8];  const float* b3 = (const float*)d_in[9];
    const float* W4 = (const float*)d_in[10]; const float* g4 = (const float*)d_in[11]; const float* b4 = (const float*)d_in[12];
    const float* W5 = (const float*)d_in[13]; const float* g5 = (const float*)d_in[14]; const float* b5 = (const float*)d_in[15];
    const float* W6 = (const float*)d_in[16]; const float* g6 = (const float*)d_in[17]; const float* b6 = (const float*)d_in[18];
    float* out = (float*)d_out;

    const int B = BATCH, N = NPTS, Kk = KNBR;
    const int L = N * Kk;                               // 40960 cols/batch

    // workspace layout (all sizes 256B-aligned)
    char* w = (char*)d_ws;
    const size_t szYA   = (size_t)B * 64 * L * 4;       // 167,772,160 (also y6)
    const size_t szZB   = (size_t)B * L * 64 * 2;       //  83,886,080 col-major
    const size_t szX    = (size_t)B * 64 * N * 4;       //   8,388,608
    const size_t szXcat = (size_t)B * N * 192 * 2;      //  12,582,912 col-major
    const size_t szIdx  = (size_t)B * N * Kk * 4;       //   2,621,440
    const size_t szXX   = (size_t)B * N * 4;            //     131,072
    const size_t szPart = (size_t)B * EMB * 2 * 4;      //     131,072
    const size_t szWpk  = (size_t)(EMB / 16) * 6 * 32 * 16 * 2;  // 393,216
    size_t off = 0;
    float*          yA   = (float*)(w + off);          off += szYA;
    unsigned short* zB   = (unsigned short*)(w + off); off += szZB;
    float*          x1   = (float*)(w + off);          off += szX;
    float*          x2   = (float*)(w + off);          off += szX;
    float*          x3   = (float*)(w + off);          off += szX;
    unsigned short* xcat = (unsigned short*)(w + off); off += szXcat;
    int*            idx  = (int*)(w + off);            off += szIdx;
    float*          xx   = (float*)(w + off);          off += szXX;
    float*          part = (float*)(w + off);          off += szPart;
    unsigned short* wpk  = (unsigned short*)(w + off); off += szWpk;
    float*          mu   = (float*)(w + off);          off += 4096;
    float*          rs   = (float*)(w + off);          off += 4096;
    if (ws_size < off) return;

    const int colBlocks2d = (B * L / 16) / 8;           // 5,120 (8 waves/block)
    const float invCnt2d  = 1.f / (float)((size_t)B * L);
    dim3 knnGrid(N / 128, B);

    // ---------------- EdgeConv block 1 (C=3) ----------------
    k_sqnorm<<<B * N / 256, 256, 0, stream>>>(x, xx, 3, N);
    k_knn<3><<<knnGrid, 128, 0, stream>>>(x, xx, idx, N, Kk);
    k_wpack<<<8, 256, 0, stream>>>(W1, wpk, 6, 64);
    k_conv_gf_t<3, 64, NPTS><<<colBlocks2d, 256, 0, stream>>>(x, idx, wpk, yA);
    k_stats_part<<<B * 64, 256, 0, stream>>>(yA, part, L);
    k_stats_fin<<<1, 256, 0, stream>>>(part, mu, rs, 64, B, invCnt2d);
    k_bn_tr<64><<<B * L / 256, 256, 0, stream>>>(yA, zB, mu, rs, g1, b1, L);
    k_wpack<<<16, 256, 0, stream>>>(W2, wpk, 64, 64);
    k_conv_bf_t<64, 64, NPTS * KNBR><<<colBlocks2d, 256, 0, stream>>>(zB, wpk, yA);
    k_stats_part<<<B * 64, 256, 0, stream>>>(yA, part, L);
    k_stats_fin<<<1, 256, 0, stream>>>(part, mu, rs, 64, B, invCnt2d);
    k_bnmax<<<B * 64 * N / 256, 256, 0, stream>>>(yA, x1, xcat, mu, rs, g2, b2, 64, N, Kk, 192, 0);

    // ---------------- EdgeConv block 2 (C=64) ----------------
    k_sqnorm<<<B * N / 256, 256, 0, stream>>>(x1, xx, 64, N);
    k_knn<64><<<knnGrid, 128, 0, stream>>>(x1, xx, idx, N, Kk);
    k_wpack<<<32, 256, 0, stream>>>(W3, wpk, 128, 64);
    k_conv_gf_t<64, 64, NPTS><<<colBlocks2d, 256, 0, stream>>>(x1, idx, wpk, yA);
    k_stats_part<<<B * 64, 256, 0, stream>>>(yA, part, L);
    k_stats_fin<<<1, 256, 0, stream>>>(part, mu, rs, 64, B, invCnt2d);
    k_bn_tr<64><<<B * L / 256, 256, 0, stream>>>(yA, zB, mu, rs, g3, b3, L);
    k_wpack<<<16, 256, 0, stream>>>(W4, wpk, 64, 64);
    k_conv_bf_t<64, 64, NPTS * KNBR><<<colBlocks2d, 256, 0, stream>>>(zB, wpk, yA);
    k_stats_part<<<B * 64, 256, 0, stream>>>(yA, part, L);
    k_stats_fin<<<1, 256, 0, stream>>>(part, mu, rs, 64, B, invCnt2d);
    k_bnmax<<<B * 64 * N / 256, 256, 0, stream>>>(yA, x2, xcat, mu, rs, g4, b4, 64, N, Kk, 192, 64);

    // ---------------- EdgeConv block 3 (C=64) ----------------
    k_sqnorm<<<B * N / 256, 256, 0, stream>>>(x2, xx, 64, N);
    k_knn<64><<<knnGrid, 128, 0, stream>>>(x2, xx, idx, N, Kk);
    k_wpack<<<32, 256, 0, stream>>>(W5, wpk, 128, 64);
    k_conv_gf_t<64, 64, NPTS><<<colBlocks2d, 256, 0, stream>>>(x2, idx, wpk, yA);
    k_stats_part<<<B * 64, 256, 0, stream>>>(yA, part, L);
    k_stats_fin<<<1, 256, 0, stream>>>(part, mu, rs, 64, B, invCnt2d);
    k_bnmax<<<B * 64 * N / 256, 256, 0, stream>>>(yA, x3, xcat, mu, rs, g5, b5, 64, N, Kk, 192, 128);

    // ---------------- final 192 -> 1024 conv ----------------
    k_wpack<<<768, 256, 0, stream>>>(W6, wpk, 192, EMB);
    k_conv_bf_t<192, EMB, NPTS><<<(B * N / 16) / 8, 256, 0, stream>>>(xcat, wpk, yA);
    k_stats_part<<<B * EMB, 256, 0, stream>>>(yA, part, N);
    k_stats_fin<<<EMB / 256, 256, 0, stream>>>(part, mu, rs, EMB, B, 1.f / (float)((size_t)B * N));
    k_final_t<<<dim3(N / 64, EMB / 64, B), 256, 0, stream>>>(yA, out, mu, rs, g6, b6, EMB, N);
}
